// DenseEncoder_88785563943194
// MI455X (gfx1250) — matrix operand order repaired and checked by reading, not compiled, not run on hardware
//
#include <hip/hip_runtime.h>
#include <hip/hip_bf16.h>

typedef unsigned short u16;
typedef unsigned int   u32;
typedef __attribute__((ext_vector_type(16))) __bf16 v16bf;
typedef __attribute__((ext_vector_type(8)))  float  v8f;
typedef __attribute__((ext_vector_type(4)))  u32    v4u;
typedef __attribute__((ext_vector_type(4)))  int    v4i;

union Frag { v16bf v; v4u q[2]; };

#if defined(__has_builtin)
# if __has_builtin(__builtin_amdgcn_global_load_async_to_lds_b128)
#  define USE_ASYNC_LDS 1
# endif
#endif
#ifndef USE_ASYNC_LDS
# define USE_ASYNC_LDS 0
#endif

// builtin signature (from hipcc diagnostic): (v4i* gsrc, v4i AS3* ldst, imm, imm)
#define ASYNC_CP_B128(gp, lp)                                            \
  __builtin_amdgcn_global_load_async_to_lds_b128(                        \
      (v4i*)(gp), (__attribute__((address_space(3))) v4i*)(lp), 0, 0)

__device__ __forceinline__ void wait_async0() {
#if defined(__has_builtin) && __has_builtin(__builtin_amdgcn_s_wait_asynccnt)
  __builtin_amdgcn_s_wait_asynccnt(0);
#else
  asm volatile("s_wait_asynccnt 0x0" ::: "memory");
#endif
}

__device__ __forceinline__ u16 f2bf(float f) {
  union { float f; u32 u; } x; x.f = f;
  u32 r = x.u + 0x7fffu + ((x.u >> 16) & 1u);
  return (u16)(r >> 16);
}
__device__ __forceinline__ float bf2f(u16 h) {
  union { u32 u; float f; } x; x.u = ((u32)h) << 16;
  return x.f;
}

// ---------------------------------------------------------------------------
// Implicit-GEMM 3D conv on v_wmma_f32_16x16x32_bf16.
//   Block tile: 128 M (output voxels) x NGRP*16 N (cout).
//   Wave tile : 32 M (2 A-frags) x NGRP*16 N  -> 2*NGRP WMMA per K-chunk.
//   K        : taps * (CIN_S/32) chunks of 32 bf16, double-buffered in LDS,
//              prefetch of chunk k+1 (async-to-LDS when available) overlaps
//              WMMA of chunk k.
// Activations: NHWC bf16, channels padded to CIN_S. Weights: [tap][COUT][CIN_S].
// Epilogue: bf16 pre-BN store + per-channel f32 sum/sum^2 atomics (training BN).
// ---------------------------------------------------------------------------
template<int CIN_S,int COUT,int NGRP,int COUT_S,
         int KD,int KH,int KW,int PD,int PH,int PW,int SW,
         int ID,int IH,int IW,int OD,int OH,int OW>
__global__ __launch_bounds__(128)
void conv_wmma(const u16* __restrict__ act_in, const u16* __restrict__ wgt,
               u16* __restrict__ act_out, float* __restrict__ sum_,
               float* __restrict__ sq_)
{
  constexpr int MTOT = OD * OH * OW;
  constexpr int TAPS = KD * KH * KW;
  constexpr int NCK  = CIN_S / 32;
  constexpr int KCH  = TAPS * NCK;
  constexpr int NR   = NGRP * 16;

  __shared__ alignas(16) u16 As[2][128][32];
  __shared__ alignas(16) u16 Bs[2][NR][32];

  const int tid   = threadIdx.x;
  const int lane  = tid & 31;
  const int wave  = tid >> 5;
  const int mbase = blockIdx.x * 128;
  const int nbase = blockIdx.y * NR;

  v8f acc[2][NGRP];
#pragma unroll
  for (int t = 0; t < 2; ++t)
#pragma unroll
    for (int g = 0; g < NGRP; ++g)
      acc[t][g] = v8f{0.f,0.f,0.f,0.f,0.f,0.f,0.f,0.f};

  // A loader: one 64B activation row per thread
  const int am  = mbase + tid;
  const bool amv = am < MTOT;
  int ax0 = 0, ax1 = 0, ax2 = 0;
  if (amv) {
    ax0 = am / (OH * OW);
    int r = am - ax0 * (OH * OW);
    ax1 = r / OW;
    ax2 = r - ax1 * OW;
  }
  // B loader: 2 threads per weight row
  const int  brow = tid >> 1;
  const int  bseg = tid & 1;
  const bool bact = brow < NR;
  const int  bn   = nbase + brow;

  // WMMA fragment gather offsets (halves) per ISA VGPR layouts
  const int ka  = (lane < 16) ? 0 : 8;    // A: K {0-7,16-23} / {8-15,24-31}
  const int kb  = (lane < 16) ? 0 : 8;    // B: K {0-7,16-23} / {8-15,24-31}
  const int r0a = (wave << 5) + (lane & 15);
  const int r0b = r0a + 16;

  auto chunk_geom = [&](int k, bool& inb, size_t& ab, size_t& bb) {
    const int tap = k / NCK, cb = k - tap * NCK;
    const int kw = tap % KW;
    const int kh = (tap / KW) % KH;
    const int kd = tap / (KW * KH);
    const int i0 = ax0 + kd - PD;
    const int i1 = ax1 + kh - PH;
    const int i2 = ax2 * SW + kw - PW;
    inb = amv && (unsigned)i0 < (unsigned)ID && (unsigned)i1 < (unsigned)IH &&
          (unsigned)i2 < (unsigned)IW;
    ab = inb ? ((size_t)((i0 * IH + i1) * IW + i2)) * CIN_S + cb * 32 : 0;
    bb = ((size_t)tap * COUT + bn) * CIN_S + cb * 32 + bseg * 16;
  };

  auto compute = [&](int s) {
    Frag fa0, fa1;
    fa0.q[0] = *(const v4u*)&As[s][r0a][ka];
    fa0.q[1] = *(const v4u*)&As[s][r0a][ka + 16];
    fa1.q[0] = *(const v4u*)&As[s][r0b][ka];
    fa1.q[1] = *(const v4u*)&As[s][r0b][ka + 16];
#pragma unroll
    for (int g = 0; g < NGRP; ++g) {
      Frag fb;
      const int rb_ = (g << 4) + (lane & 15);
      fb.q[0] = *(const v4u*)&Bs[s][rb_][kb];
      fb.q[1] = *(const v4u*)&Bs[s][rb_][kb + 16];
      acc[0][g] = __builtin_amdgcn_wmma_f32_16x16x32_bf16(
          false, fa0.v, false, fb.v, (short)0, acc[0][g], false, false);
      acc[1][g] = __builtin_amdgcn_wmma_f32_16x16x32_bf16(
          false, fa1.v, false, fb.v, (short)0, acc[1][g], false, false);
    }
  };

#if USE_ASYNC_LDS
  auto load_chunk = [&](int k, int s) {
    bool inb; size_t ab, bb;
    chunk_geom(k, inb, ab, bb);
    u16* la = &As[s][tid][0];
    if (inb) {
      const u16* ga = act_in + ab;
#pragma unroll
      for (int q = 0; q < 4; ++q)
        ASYNC_CP_B128(ga + q * 8, la + q * 8);
    } else {
      const v4u z = {0, 0, 0, 0};
#pragma unroll
      for (int q = 0; q < 4; ++q) *(v4u*)(la + q * 8) = z;
    }
    if (bact) {
      const u16* gb = wgt + bb;
      u16* lb = &Bs[s][brow][bseg * 16];
      ASYNC_CP_B128(gb, lb);
      ASYNC_CP_B128(gb + 8, lb + 8);
    }
  };

  load_chunk(0, 0);
#pragma unroll 1
  for (int k = 0; k < KCH; ++k) {
    wait_async0();          // chunk k resident in LDS (this wave's requests)
    __syncthreads();        // ... and everyone else's
    if (k + 1 < KCH) load_chunk(k + 1, (k + 1) & 1);
    compute(k & 1);
  }
#else
  v4u ra0, ra1, ra2, ra3, rb0, rb1;
  auto gload = [&](int k, v4u& a0, v4u& a1, v4u& a2, v4u& a3, v4u& b0, v4u& b1) {
    bool inb; size_t ab, bb;
    chunk_geom(k, inb, ab, bb);
    a0 = a1 = a2 = a3 = v4u{0, 0, 0, 0};
    b0 = b1 = v4u{0, 0, 0, 0};
    if (inb) {
      const v4u* p = (const v4u*)(act_in + ab);
      a0 = p[0]; a1 = p[1]; a2 = p[2]; a3 = p[3];
    }
    if (bact) {
      const v4u* q = (const v4u*)(wgt + bb);
      b0 = q[0]; b1 = q[1];
    }
  };
  auto sstore = [&](int s, v4u a0, v4u a1, v4u a2, v4u a3, v4u b0, v4u b1) {
    v4u* la = (v4u*)&As[s][tid][0];
    la[0] = a0; la[1] = a1; la[2] = a2; la[3] = a3;
    if (bact) {
      v4u* lb = (v4u*)&Bs[s][brow][bseg * 16];
      lb[0] = b0; lb[1] = b1;
    }
  };

  gload(0, ra0, ra1, ra2, ra3, rb0, rb1);
#pragma unroll 1
  for (int k = 0; k < KCH; ++k) {
    sstore(k & 1, ra0, ra1, ra2, ra3, rb0, rb1);
    if (k + 1 < KCH) gload(k + 1, ra0, ra1, ra2, ra3, rb0, rb1); // overlaps compute
    __syncthreads();
    compute(k & 1);
  }
#endif

  // Epilogue: bf16 pre-BN activations + per-channel BN statistics.
  const int nloc  = lane & 15;
  const int mrow0 = (lane < 16) ? 0 : 8;
#pragma unroll
  for (int t = 0; t < 2; ++t) {
#pragma unroll
    for (int g = 0; g < NGRP; ++g) {
      const int n = nbase + (g << 4) + nloc;   // always < COUT by launch config
      float s = 0.f, q = 0.f;
#pragma unroll
      for (int j = 0; j < 8; ++j) {
        const float v = acc[t][g][j];
        s += v; q += v * v;
        const int m = mbase + (wave << 5) + t * 16 + mrow0 + j;
        if (m < MTOT) act_out[(size_t)m * COUT_S + n] = f2bf(v);
      }
      atomicAdd(&sum_[n], s);
      atomicAdd(&sq_[n],  q);
    }
  }
}

// ---------------------------------------------------------------------------
__global__ void zero_b128(v4u* __restrict__ p, size_t n) {
  const size_t stride = (size_t)gridDim.x * blockDim.x;
  for (size_t i = (size_t)blockIdx.x * blockDim.x + threadIdx.x; i < n; i += stride)
    p[i] = v4u{0,0,0,0};
}

__global__ void wcvt(const float* __restrict__ W, u16* __restrict__ dst,
                     int cout, int cin, int cin_s, int KDv, int KHv, int KWv) {
  const size_t total  = (size_t)KDv * KHv * KWv * cout * cin_s;
  const size_t stride = (size_t)gridDim.x * blockDim.x;
  for (size_t i = (size_t)blockIdx.x * blockDim.x + threadIdx.x; i < total; i += stride) {
    const int ci  = (int)(i % cin_s);
    const size_t t = i / cin_s;
    const int o   = (int)(t % cout);
    const int tap = (int)(t / cout);
    const int kw = tap % KWv;
    const int kh = (tap / KWv) % KHv;
    const int kd = tap / (KWv * KHv);
    float v = 0.f;
    if (ci < cin)
      v = W[((((size_t)o * cin + ci) * KDv + kd) * KHv + kh) * KWv + kw];
    dst[i] = f2bf(v);
  }
}

__global__ void scatter_vox(const float* __restrict__ vf, const int* __restrict__ coors,
                            u16* __restrict__ act, int N) {
  const int i = blockIdx.x * blockDim.x + threadIdx.x;
  if (i < N) {
    const int x = coors[i * 4 + 1];
    const int y = coors[i * 4 + 2];
    const int z = coors[i * 4 + 3];
    const size_t base = ((size_t)((x * 128 + y) * 41 + z)) * 32;
#pragma unroll
    for (int c = 0; c < 5; ++c) act[base + c] = f2bf(vf[i * 5 + c]);
  }
}

__global__ void bn_stats(const float* __restrict__ sum_, const float* __restrict__ sq_,
                         const float* __restrict__ g, const float* __restrict__ b,
                         float* __restrict__ scale, float* __restrict__ shift,
                         int C, float invcnt) {
  const int c = blockIdx.x * blockDim.x + threadIdx.x;
  if (c < C) {
    const float m   = sum_[c] * invcnt;
    const float var = sq_[c] * invcnt - m * m;
    const float inv = rsqrtf(var + 1e-3f);
    const float sc  = g[c] * inv;
    scale[c] = sc;
    shift[c] = b[c] - m * sc;
  }
}

__global__ void bn_apply(u16* __restrict__ buf, const float* __restrict__ scale,
                         const float* __restrict__ shift, int Mtot, int C, int Cs) {
  const size_t total  = (size_t)Mtot * Cs;
  const size_t stride = (size_t)gridDim.x * blockDim.x;
  for (size_t i = (size_t)blockIdx.x * blockDim.x + threadIdx.x; i < total; i += stride) {
    const int c = (int)(i % Cs);
    if (c < C) {
      const float x = bf2f(buf[i]);
      const float y = fmaxf(x * scale[c] + shift[c], 0.f);
      buf[i] = f2bf(y);
    } else {
      buf[i] = 0;
    }
  }
}

// Final BN+ReLU + permute [126,128,20,128] -> out[c*20+z][x0][x1] (f32)
__global__ void bn_apply_final(const u16* __restrict__ buf, const float* __restrict__ scale,
                               const float* __restrict__ shift, float* __restrict__ out) {
  constexpr int M = 126 * 128 * 20;
  const size_t total  = (size_t)M * 128;
  const size_t stride = (size_t)gridDim.x * blockDim.x;
  for (size_t i = (size_t)blockIdx.x * blockDim.x + threadIdx.x; i < total; i += stride) {
    const int c  = (int)(i & 127);
    const size_t m = i >> 7;
    const int x0 = (int)(m / (128 * 20));
    const int r  = (int)(m % (128 * 20));
    const int x1 = r / 20;
    const int z  = r % 20;
    const float y = fmaxf(bf2f(buf[i]) * scale[c] + shift[c], 0.f);
    out[(((size_t)c * 20 + z) * 126 + x0) * 128 + x1] = y;
  }
}

// ---------------------------------------------------------------------------
extern "C" void kernel_launch(void* const* d_in, const int* in_sizes, int n_in,
                              void* d_out, int out_size, void* d_ws, size_t ws_size,
                              hipStream_t stream) {
  (void)n_in; (void)out_size; (void)ws_size;

  const float* vf    = (const float*)d_in[0];
  const int*   coors = (const int*)d_in[1];
  const int    Nvox  = in_sizes[0] / 5;

  static const int LCOUT[11] = {16,32,32,32,64,64,64,64,64,64,128};
  static const int LCIN [11] = { 5,16,32,32,32,64,64,64,64,64, 64};
  static const int LCINS[11] = {32,32,32,32,32,64,64,64,64,64, 64};
  static const int LKD  [11] = { 3, 3, 3, 3, 3, 3, 3, 3, 3, 3,  1};
  static const int LKH  [11] = { 3, 3, 3, 3, 3, 3, 3, 3, 3, 3,  1};
  static const int LKW  [11] = { 3, 3, 3, 3, 3, 3, 3, 3, 3, 3,  3};

  constexpr size_t ACT_ELEMS = 671744ull * 64;   // 128*128*41 voxels * 64ch, bf16
  u16*   act0  = (u16*)d_ws;
  u16*   act1  = act0 + ACT_ELEMS;
  float* stats = (float*)(act1 + ACT_ELEMS);     // 512 floats per layer
  u16*   wgt   = (u16*)(stats + 512 * 11);

  size_t woff[12];
  woff[0] = 0;
  for (int l = 0; l < 11; ++l)
    woff[l + 1] = woff[l] + (size_t)LKD[l] * LKH[l] * LKW[l] * LCOUT[l] * LCINS[l];

  auto SUM = [&](int l) { return stats + 512 * l; };
  auto SQ  = [&](int l) { return stats + 512 * l + 128; };
  auto SCL = [&](int l) { return stats + 512 * l + 256; };
  auto SHF = [&](int l) { return stats + 512 * l + 384; };
  auto Wp  = [&](int l) { return (const float*)d_in[3 + 3 * l]; };
  auto Gp  = [&](int l) { return (const float*)d_in[4 + 3 * l]; };
  auto Bp  = [&](int l) { return (const float*)d_in[5 + 3 * l]; };

  // 1) zero BN stats + layer-0 input arena (deterministic per launch)
  {
    size_t n4 = (512ull * 11 * 4) / 16;
    zero_b128<<<(unsigned)((n4 + 255) / 256), 256, 0, stream>>>((v4u*)stats, n4);
    size_t a4 = (671744ull * 32 * 2) / 16;
    zero_b128<<<(unsigned)((a4 + 255) / 256), 256, 0, stream>>>((v4u*)act0, a4);
  }

  // 2) convert weights f32 OIDHW -> bf16 [tap][cout][cin_s]
  for (int l = 0; l < 11; ++l) {
    const size_t total = woff[l + 1] - woff[l];
    wcvt<<<(unsigned)((total + 255) / 256), 256, 0, stream>>>(
        Wp(l), wgt + woff[l], LCOUT[l], LCIN[l], LCINS[l], LKD[l], LKH[l], LKW[l]);
  }

  // 3) scatter voxel features into dense bf16 grid (ch padded to 32)
  scatter_vox<<<(Nvox + 255) / 256, 256, 0, stream>>>(vf, coors, act0, Nvox);

  constexpr int M128 = 128 * 128 * 41;  // 671744 = 5248*128
  constexpr int M126 = 126 * 128 * 41;  // 661248 = 5166*128
  constexpr int MFIN = 126 * 128 * 20;  // 322560 = 2520*128
  const float i128 = 1.0f / (float)M128;
  const float i126 = 1.0f / (float)M126;
  const float ifin = 1.0f / (float)MFIN;

  // ------ L0: 5(pad32) -> 16(pad32), act0 -> act1 ------
  conv_wmma<32,16,1,32, 3,3,3, 1,1,1, 1, 128,128,41, 128,128,41>
      <<<dim3(M128/128,1), 128, 0, stream>>>(act0, wgt+woff[0], act1, SUM(0), SQ(0));
  bn_stats<<<1,128,0,stream>>>(SUM(0),SQ(0),Gp(0),Bp(0),SCL(0),SHF(0),16,i128);
  bn_apply<<<(unsigned)(((size_t)M128*32+255)/256),256,0,stream>>>(act1,SCL(0),SHF(0),M128,16,32);

  // ------ L1..L3: 32 -> 32 ------
  u16* cin_buf = act1; u16* cout_buf = act0;
  for (int l = 1; l <= 3; ++l) {
    conv_wmma<32,32,2,32, 3,3,3, 1,1,1, 1, 128,128,41, 128,128,41>
        <<<dim3(M128/128,1), 128, 0, stream>>>(cin_buf, wgt+woff[l], cout_buf, SUM(l), SQ(l));
    bn_stats<<<1,128,0,stream>>>(SUM(l),SQ(l),Gp(l),Bp(l),SCL(l),SHF(l),32,i128);
    bn_apply<<<(unsigned)(((size_t)M128*32+255)/256),256,0,stream>>>(cout_buf,SCL(l),SHF(l),M128,32,32);
    u16* t = cin_buf; cin_buf = cout_buf; cout_buf = t;
  }
  // after L3: activations in act0

  // ------ L4: 32 -> 64, act0 -> act1 ------
  conv_wmma<32,64,4,64, 3,3,3, 1,1,1, 1, 128,128,41, 128,128,41>
      <<<dim3(M128/128,1), 128, 0, stream>>>(act0, wgt+woff[4], act1, SUM(4), SQ(4));
  bn_stats<<<1,128,0,stream>>>(SUM(4),SQ(4),Gp(4),Bp(4),SCL(4),SHF(4),64,i128);
  bn_apply<<<(unsigned)(((size_t)M128*64+255)/256),256,0,stream>>>(act1,SCL(4),SHF(4),M128,64,64);

  // ------ L5, L6: 64 -> 64, 128-depth ------
  cin_buf = act1; cout_buf = act0;
  for (int l = 5; l <= 6; ++l) {
    conv_wmma<64,64,4,64, 3,3,3, 1,1,1, 1, 128,128,41, 128,128,41>
        <<<dim3(M128/128,1), 128, 0, stream>>>(cin_buf, wgt+woff[l], cout_buf, SUM(l), SQ(l));
    bn_stats<<<1,128,0,stream>>>(SUM(l),SQ(l),Gp(l),Bp(l),SCL(l),SHF(l),64,i128);
    bn_apply<<<(unsigned)(((size_t)M128*64+255)/256),256,0,stream>>>(cout_buf,SCL(l),SHF(l),M128,64,64);
    u16* t = cin_buf; cin_buf = cout_buf; cout_buf = t;
  }
  // after L6: activations in act1

  // ------ L7: 64 -> 64, padding (0,1,1): depth 128 -> 126, act1 -> act0 ------
  conv_wmma<64,64,4,64, 3,3,3, 0,1,1, 1, 128,128,41, 126,128,41>
      <<<dim3(M126/128,1), 128, 0, stream>>>(act1, wgt+woff[7], act0, SUM(7), SQ(7));
  bn_stats<<<1,128,0,stream>>>(SUM(7),SQ(7),Gp(7),Bp(7),SCL(7),SHF(7),64,i126);
  bn_apply<<<(unsigned)(((size_t)M126*64+255)/256),256,0,stream>>>(act0,SCL(7),SHF(7),M126,64,64);

  // ------ L8, L9: 64 -> 64 at 126-depth ------
  cin_buf = act0; cout_buf = act1;
  for (int l = 8; l <= 9; ++l) {
    conv_wmma<64,64,4,64, 3,3,3, 1,1,1, 1, 126,128,41, 126,128,41>
        <<<dim3(M126/128,1), 128, 0, stream>>>(cin_buf, wgt+woff[l], cout_buf, SUM(l), SQ(l));
    bn_stats<<<1,128,0,stream>>>(SUM(l),SQ(l),Gp(l),Bp(l),SCL(l),SHF(l),64,i126);
    bn_apply<<<(unsigned)(((size_t)M126*64+255)/256),256,0,stream>>>(cout_buf,SCL(l),SHF(l),M126,64,64);
    u16* t = cin_buf; cin_buf = cout_buf; cout_buf = t;
  }
  // after L9: activations in act0

  // ------ conv_out: 64 -> 128, kernel (1,1,3), stride (1,1,2), act0 -> act1 ------
  conv_wmma<64,128,4,128, 1,1,3, 0,0,0, 2, 126,128,41, 126,128,20>
      <<<dim3(MFIN/128,2), 128, 0, stream>>>(act0, wgt+woff[10], act1, SUM(10), SQ(10));
  bn_stats<<<1,128,0,stream>>>(SUM(10),SQ(10),Gp(10),Bp(10),SCL(10),SHF(10),128,ifin);
  bn_apply_final<<<(unsigned)(((size_t)MFIN*128+255)/256),256,0,stream>>>(
      act1, SCL(10), SHF(10), (float*)d_out);
}
